// LocalAutoregressiveHead_48739288875014
// MI455X (gfx1250) — matrix operand order repaired and checked by reading, not compiled, not run on hardware
//
#include <hip/hip_runtime.h>
#include <hip/hip_bf16.h>

// ---------------------------------------------------------------------------
// LocalAutoregressiveHead on MI455X (gfx1250, wave32, WMMA bf16 w/ f32 accum)
//
// Pipeline (all on `stream`):
//   K1 prep_convert : f32->bf16 weight copies + SOS-shifted byte table
//   K2 embtab       : embtab[v,g] = emb_table[v]·W_ih[g,:256] + b_ih + b_hh(r,z)
//   K3 latent_proj  : latent_ctx(bf16) = latents @ W_proj^T + b_proj   (WMMA)
//   K4 gru_head     : per 64-row tile: lat_gate->LDS (WMMA), 16-step GRU
//                     (WMMA h@W_hh^T) + fused head GEMM (WMMA) -> logits
//
// B-fragment streams use a ping-pong pair bq[0]/bq[1] (compile-time parity):
// the loads for tile i+1 target bq[(i+1)&1] and issue before the v_wmma that
// consumes bq[i&1] -- overlap without register-rotation copies.
// ---------------------------------------------------------------------------

typedef __attribute__((ext_vector_type(16))) __bf16       bf16x16;
typedef __attribute__((ext_vector_type(8)))  float        f32x8;
typedef __attribute__((ext_vector_type(4)))  float        f32x4;
typedef __attribute__((ext_vector_type(4)))  unsigned int u32x4;

union FragB16 { bf16x16 v; u32x4 u[2]; };

#define D_MODEL 1024
#define HID     256
#define G3      768
#define PATCH   16
#define VOCAB   256
#define BN      8192
#define TILE_M  64

// A-matrix 16x32 bf16 frag (ISA 7.12.2): lane(row m=lane&15, half=lane>>4)
// holds K = {k0+8h .. +7} and {k0+16+8h .. +7}  -> two b128 loads.
__device__ __forceinline__ bf16x16 load_frag_a(const __bf16* base, int lda,
                                               int row0, int k0, int lane) {
  const int h = lane >> 4, m = lane & 15;
  const __bf16* p = base + (size_t)(row0 + m) * lda + k0 + 8 * h;
  FragB16 f;
  f.u[0] = *(const u32x4*)p;
  f.u[1] = *(const u32x4*)(p + 16);
  return f.v;
}

// B-matrix 32x16 bf16 frag: lane(col n=lane&15, half=lane>>4) holds
// K = {k0+8h .. +7} and {k0+16+8h .. +7} of column n. B[k,n] = W[n,k] (W row-major, ld=K).
__device__ __forceinline__ bf16x16 load_frag_b(const __bf16* base, int ldb,
                                               int n0, int k0, int lane) {
  const int h = lane >> 4, n = lane & 15;
  const __bf16* p = base + (size_t)(n0 + n) * ldb + k0 + 8 * h;
  FragB16 f;
  f.u[0] = *(const u32x4*)p;
  f.u[1] = *(const u32x4*)(p + 16);
  return f.v;
}

__device__ __forceinline__ f32x8 wmma_bf16(bf16x16 a, bf16x16 b, f32x8 c) {
  return __builtin_amdgcn_wmma_f32_16x16x32_bf16(false, a, false, b,
                                                 (short)0, c, false, false);
}

__device__ __forceinline__ float sigmoidf_(float x) {
  return 1.0f / (1.0f + __expf(-x));
}

// acc[3][4] += A(16 rows at arow0, K=256) x Bw^T; 12 B tiles = 3 gates x this
// wave's 4 col-tiles. Ping-pong B frags, fully unrolled (parity continuous:
// 12 is even, so bq[0] always feeds tile 0 of the next k-step).
__device__ __forceinline__ void gemm_12tile(const __bf16* __restrict__ Bw,
                                            const __bf16* Abase, int lda,
                                            int arow0, int cg, int lane,
                                            f32x8 (&acc)[3][4]) {
  bf16x16 bq[2];
  bq[0] = load_frag_b(Bw, HID, cg * 64, 0, lane);
#pragma unroll
  for (int kk = 0; kk < 8; ++kk) {
    bf16x16 a = load_frag_a(Abase, lda, arow0, kk * 32, lane);
#pragma unroll
    for (int i = 0; i < 12; ++i) {
      const int nx = i + 1;
      if (!(kk == 7 && nx == 12)) {     // compile-time guard (no implicit-def)
        const int ni = nx % 12;
        const int nkk = (nx == 12) ? kk + 1 : kk;
        bq[nx & 1] = load_frag_b(
            Bw, HID, (ni >> 2) * HID + (cg * 4 + (ni & 3)) * 16, nkk * 32,
            lane);
      }
      acc[i >> 2][i & 3] = wmma_bf16(a, bq[i & 1], acc[i >> 2][i & 3]);
    }
  }
}

// ---------------------------------------------------------------- K1: prep
__global__ __launch_bounds__(256) void prep_convert(
    const float* __restrict__ wproj, const float* __restrict__ wih,
    const float* __restrict__ whh,   const float* __restrict__ whead,
    const int*   __restrict__ tb,
    __bf16* __restrict__ wproj_bf, __bf16* __restrict__ wu_bf,
    __bf16* __restrict__ whh_bf,   __bf16* __restrict__ whead_bf,
    int* __restrict__ bytes_sh) {
  const int i = blockIdx.x * 256 + threadIdx.x;
  if (i < HID * D_MODEL) wproj_bf[i] = (__bf16)wproj[i];
  if (i < G3 * HID) {
    const int r = i >> 8, c = i & 255;
    wu_bf[i]  = (__bf16)wih[(size_t)r * (2 * HID) + HID + c];  // latent half
    whh_bf[i] = (__bf16)whh[i];
  }
  if (i < VOCAB * HID) whead_bf[i] = (__bf16)whead[i];
  if (i < BN * PATCH) {               // SOS-shifted input byte per (row, step)
    const int row = i >> 4, p = i & 15;
    bytes_sh[i] = (p == 0) ? 0 : tb[row * PATCH + p - 1];
  }
}

// ---------------------------------------------------------------- K2: embtab
// embtab[v][g] = emb_table[v]·W_ih[g, 0:256] + b_ih[g] + (g<512 ? b_hh[g] : 0)
// (b_hh of the n-gate must stay separate: it is scaled by r in the GRU.)
__global__ __launch_bounds__(768) void embtab_kernel(
    const float* __restrict__ emb, const float* __restrict__ wih,
    const float* __restrict__ b_ih, const float* __restrict__ b_hh,
    float* __restrict__ embtab) {
  const int v = blockIdx.x, g = threadIdx.x;
  float acc = b_ih[g] + (g < 2 * HID ? b_hh[g] : 0.0f);
  const float* er = emb + (size_t)v * HID;
  const float* wr = wih + (size_t)g * (2 * HID);
  for (int h = 0; h < HID; ++h) acc += er[h] * wr[h];
  embtab[(size_t)v * G3 + g] = acc;
}

// ---------------------------------------------------------------- K3: latent
// latbf[8192,256] = bf16( latents[8192,1024] @ W_proj^T + b_proj )
__global__ __launch_bounds__(512) void latent_proj_kernel(
    const float* __restrict__ latents, const __bf16* __restrict__ wproj,
    const float* __restrict__ b_proj, __bf16* __restrict__ latbf) {
  const int tid = threadIdx.x, lane = tid & 31, wave = tid >> 5;
  const int rt = wave >> 2, cg = wave & 3;
  const int h = lane >> 4, ln = lane & 15;
  const int r0 = blockIdx.x * TILE_M + rt * 16;

  f32x8 acc[4] = {};
  bf16x16 bq[2];
  bq[0] = load_frag_b(wproj, D_MODEL, cg * 64, 0, lane);
  for (int kk = 0; kk < 32; ++kk) {
    // build A frag: convert f32 -> bf16 on the fly (tf32-style path)
    const float* pa = latents + (size_t)(r0 + ln) * D_MODEL + kk * 32 + 8 * h;
    f32x4 a0 = *(const f32x4*)(pa + 0);
    f32x4 a1 = *(const f32x4*)(pa + 4);
    f32x4 a2 = *(const f32x4*)(pa + 16);
    f32x4 a3 = *(const f32x4*)(pa + 20);
    bf16x16 a;
#pragma unroll
    for (int e = 0; e < 4; ++e) {
      a[e]      = (__bf16)a0[e];
      a[4 + e]  = (__bf16)a1[e];
      a[8 + e]  = (__bf16)a2[e];
      a[12 + e] = (__bf16)a3[e];
    }
#pragma unroll
    for (int t = 0; t < 4; ++t) {
      const int nx = t + 1;
      const int ni = nx & 3;
      int nkk = (nx == 4) ? kk + 1 : kk;
      if (nkk > 31) nkk = 31;  // clamp: one redundant unused reload, no branch
      bq[nx & 1] =
          load_frag_b(wproj, D_MODEL, (cg * 4 + ni) * 16, nkk * 32, lane);
      acc[t] = wmma_bf16(a, bq[t & 1], acc[t]);
    }
  }
#pragma unroll
  for (int t = 0; t < 4; ++t) {
    const int c = (cg * 4 + t) * 16 + ln;
    const float bp = b_proj[c];
#pragma unroll
    for (int j = 0; j < 8; ++j)
      latbf[(size_t)(r0 + j + 8 * h) * HID + c] = (__bf16)(acc[t][j] + bp);
  }
}

// ---------------------------------------------------------------- K4: GRU
__global__ __launch_bounds__(512) void gru_head_kernel(
    const __bf16* __restrict__ latbf, const __bf16* __restrict__ wu,
    const __bf16* __restrict__ whh,   const __bf16* __restrict__ whead,
    const float* __restrict__ embtab, const int* __restrict__ bytes_sh,
    const float* __restrict__ b_hh,   const float* __restrict__ b_head,
    float* __restrict__ out) {
  // 192K lat_gate + 66K double-buffered h + 4K bytes = 262K of the 320K WGP LDS
  __shared__ __align__(16) float  lat[TILE_M][G3];
  __shared__ __align__(16) __bf16 hbuf[2][TILE_M][HID + 8];  // pad: no bank dup
  __shared__ int bytes_s[TILE_M][PATCH];

  const int tid = threadIdx.x, lane = tid & 31, wave = tid >> 5;
  const int rt = wave >> 2, cg = wave & 3;        // 4 row-tiles x 4 col-groups
  const int h = lane >> 4, ln = lane & 15;
  const int r0 = blockIdx.x * TILE_M;
  const int HLD = HID + 8;

  // warm L2/L0 with the per-step streamed weights (global_prefetch path)
  for (size_t off = (size_t)tid * 128; off < (size_t)G3 * HID * 2;
       off += 512 * 128)
    __builtin_prefetch((const char*)whh + off, 0, 3);
  for (size_t off = (size_t)tid * 128; off < (size_t)VOCAB * HID * 2;
       off += 512 * 128)
    __builtin_prefetch((const char*)whead + off, 0, 3);

  for (int i = tid; i < TILE_M * PATCH; i += 512)
    bytes_s[i >> 4][i & 15] = bytes_sh[(r0 + (i >> 4)) * PATCH + (i & 15)];
  for (int i = tid; i < TILE_M * HLD; i += 512)
    hbuf[0][i / HLD][i % HLD] = (__bf16)0.0f;     // h0 = 0

  // ---- phase 1: lat_gate = latent_ctx @ W_u^T  -> LDS (reused 16 steps) ----
  {
    f32x8 acc[3][4];
#pragma unroll
    for (int g = 0; g < 3; ++g)
#pragma unroll
      for (int t = 0; t < 4; ++t) acc[g][t] = (f32x8){};
    gemm_12tile(wu, latbf, HID, r0 + rt * 16, cg, lane, acc);
#pragma unroll
    for (int g = 0; g < 3; ++g)
#pragma unroll
      for (int t = 0; t < 4; ++t) {
        const int n0 = g * HID + (cg * 4 + t) * 16;
#pragma unroll
        for (int j = 0; j < 8; ++j)
          lat[rt * 16 + j + 8 * h][n0 + ln] = acc[g][t][j];
      }
  }
  __syncthreads();

  // ---------------------------- 16-step scan ------------------------------
  for (int p = 0; p < PATCH; ++p) {
    const int cur = p & 1, nxt = cur ^ 1;

    // hg = h @ W_hh^T : 3 gates x 4 hidden col-tiles per wave
    f32x8 acc[3][4];
#pragma unroll
    for (int g = 0; g < 3; ++g)
#pragma unroll
      for (int t = 0; t < 4; ++t) acc[g][t] = (f32x8){};
    gemm_12tile(whh, &hbuf[cur][0][0], HLD, rt * 16, cg, lane, acc);

    // gate math; this wave owns h_new[rt rows][cg's 64 hidden cols]
#pragma unroll
    for (int t = 0; t < 4; ++t) {
      const int c = (cg * 4 + t) * 16 + ln;
      const float bn = b_hh[2 * HID + c];  // n-gate hidden bias (scaled by r)
#pragma unroll
      for (int j = 0; j < 8; ++j) {
        const int row = rt * 16 + j + 8 * h;
        const float* et = embtab + (size_t)bytes_s[row][p] * G3;
        const float xr = lat[row][c] + et[c];
        const float xz = lat[row][HID + c] + et[HID + c];
        const float xn = lat[row][2 * HID + c] + et[2 * HID + c];
        const float rr = sigmoidf_(xr + acc[0][t][j]);
        const float zz = sigmoidf_(xz + acc[1][t][j]);
        const float nn = tanhf(xn + rr * (acc[2][t][j] + bn));
        const float hp = (float)hbuf[cur][row][c];
        hbuf[nxt][row][c] = (__bf16)((1.0f - zz) * nn + zz * hp);
      }
    }
    __syncthreads();

    // fused head: logits[:,p,:] = h_new @ W_head^T + b_head -> HBM
    f32x8 hacc[4];
#pragma unroll
    for (int t = 0; t < 4; ++t) hacc[t] = (f32x8){};
    {
      bf16x16 bq[2];
      bq[0] = load_frag_b(whead, HID, cg * 64, 0, lane);
#pragma unroll
      for (int kk = 0; kk < 8; ++kk) {
        bf16x16 a = load_frag_a(&hbuf[nxt][0][0], HLD, rt * 16, kk * 32, lane);
#pragma unroll
        for (int t = 0; t < 4; ++t) {
          const int nx = t + 1;
          if (!(kk == 7 && nx == 4)) {
            const int ni = nx & 3;
            const int nkk = (nx == 4) ? kk + 1 : kk;
            bq[nx & 1] =
                load_frag_b(whead, HID, (cg * 4 + ni) * 16, nkk * 32, lane);
          }
          hacc[t] = wmma_bf16(a, bq[t & 1], hacc[t]);
        }
      }
    }
#pragma unroll
    for (int t = 0; t < 4; ++t) {
      const int v0 = (cg * 4 + t) * 16;
      const float bb2 = b_head[v0 + ln];
#pragma unroll
      for (int j = 0; j < 8; ++j) {
        const size_t row = (size_t)(r0 + rt * 16 + j + 8 * h);
        out[(row * PATCH + p) * VOCAB + v0 + ln] = hacc[t][j] + bb2;
      }
    }
    __syncthreads();
  }
}

// ---------------------------------------------------------------------------
extern "C" void kernel_launch(void* const* d_in, const int* in_sizes, int n_in,
                              void* d_out, int out_size, void* d_ws,
                              size_t ws_size, hipStream_t stream) {
  const float* latents   = (const float*)d_in[0];
  const int*   tbytes    = (const int*)d_in[1];
  const float* W_proj    = (const float*)d_in[2];
  const float* b_proj    = (const float*)d_in[3];
  const float* emb_table = (const float*)d_in[4];
  const float* W_ih      = (const float*)d_in[5];
  const float* W_hh      = (const float*)d_in[6];
  const float* b_ih      = (const float*)d_in[7];
  const float* b_hh      = (const float*)d_in[8];
  const float* W_head    = (const float*)d_in[9];
  const float* b_head    = (const float*)d_in[10];
  float* out = (float*)d_out;

  char* w = (char*)d_ws;
  __bf16* wproj_bf = (__bf16*)w; w += (size_t)HID * D_MODEL * 2;   // 512K
  __bf16* wu_bf    = (__bf16*)w; w += (size_t)G3 * HID * 2;        // 384K
  __bf16* whh_bf   = (__bf16*)w; w += (size_t)G3 * HID * 2;        // 384K
  __bf16* whead_bf = (__bf16*)w; w += (size_t)VOCAB * HID * 2;     // 128K
  float*  embtab   = (float*)w;  w += (size_t)VOCAB * G3 * 4;      // 768K
  __bf16* latbf    = (__bf16*)w; w += (size_t)BN * HID * 2;        // 4M
  int*    bytes_sh = (int*)w;                                      // 512K

  prep_convert<<<1024, 256, 0, stream>>>(W_proj, W_ih, W_hh, W_head, tbytes,
                                         wproj_bf, wu_bf, whh_bf, whead_bf,
                                         bytes_sh);
  embtab_kernel<<<VOCAB, G3, 0, stream>>>(emb_table, W_ih, b_ih, b_hh, embtab);
  latent_proj_kernel<<<BN / TILE_M, 512, 0, stream>>>(latents, wproj_bf,
                                                      b_proj, latbf);
  gru_head_kernel<<<BN / TILE_M, 512, 0, stream>>>(latbf, wu_bf, whh_bf,
                                                   whead_bf, embtab, bytes_sh,
                                                   b_hh, b_head, out);
}
